// VisionTransformerForSegmentationMultiScale_31086973288697
// MI455X (gfx1250) — matrix operand order repaired and checked by reading, not compiled, not run on hardware
//
#include <hip/hip_runtime.h>
#include <hip/hip_bf16.h>
#include <math.h>

// ---------------------------------------------------------------------------
// MI455X (gfx1250) implementation: all GEMMs + both attention matmuls via
// v_wmma_f32_16x16x32_f16 (wave32, f32 accumulate). ~137 GFLOP total, data
// fits L2 (192MB) -> compute-bound; f16 WMMA is the fast path. GEMM staging
// uses GLOBAL_LOAD_ASYNC_TO_LDS (ASYNCcnt double-buffering) when the
// toolchain exposes the gfx1250 async builtins, else plain LDS stores.
// ---------------------------------------------------------------------------

typedef __attribute__((ext_vector_type(16))) _Float16 v16h;
typedef __attribute__((ext_vector_type(8)))  _Float16 v8h;
typedef __attribute__((ext_vector_type(8)))  float    v8f;

#define B_    8
#define IMG_  256
#define E_    256
#define NH_   8
#define HD_   32
#define L_    8
#define N_    1024
#define HID_  512
#define GB_   1024
#define SCALE_ 0.17677669529663687f   // 32^-0.5
#define TOK_ROWS (B_ * N_)            // 8192

#if __has_builtin(__builtin_amdgcn_global_load_async_to_lds_b128) && \
    __has_builtin(__builtin_amdgcn_s_wait_asynccnt)
#define HAVE_ASYNC_LDS 1
#else
#define HAVE_ASYNC_LDS 0
#endif

#if HAVE_ASYNC_LDS
// From the toolchain's own diagnostic, the builtin takes
// '__attribute__((vector_size(16))) int *' params (AS1 src printed as
// '__device__', LDS dst in address_space(3), both non-const).
typedef int v4i_vs __attribute__((vector_size(16)));
typedef v4i_vs __attribute__((address_space(1)))* gas1_p;
typedef v4i_vs __attribute__((address_space(3)))* las3_p;
#endif

// stage 16 bytes global -> LDS (per-lane). Async on gfx1250 when available.
static __device__ __forceinline__ void stage16(const _Float16* g, _Float16* l) {
#if HAVE_ASYNC_LDS
  // generic global addr == AS1 addr; low 32 bits of generic LDS addr == LDS
  // offset (ISA aperture rules), AS3 pointers are 32-bit.
  __builtin_amdgcn_global_load_async_to_lds_b128(
      (gas1_p)(unsigned long long)(const void*)g,
      (las3_p)(unsigned int)(unsigned long long)(void*)l,
      /*offset=*/0, /*cpol=*/0);
#else
  *(v8h*)l = *(const v8h*)g;
#endif
}

template <int Ncnt>
static __device__ __forceinline__ void stage_wait() {
#if HAVE_ASYNC_LDS
  __builtin_amdgcn_s_wait_asynccnt(Ncnt);
#endif
}

static __device__ __forceinline__ v8f wmma16(v16h a, v16h b, v8f c) {
  // D = A(16x32 f16) * B(32x16 f16) + C(16x16 f32)
  return __builtin_amdgcn_wmma_f32_16x16x32_f16(
      /*neg_a=*/false, a, /*neg_b=*/false, b,
      /*c_mod=*/(short)0, c, /*reuse_a=*/false, /*reuse_b=*/false);
}

// A fragment: row m = lane&15; lanes 0-15 hold K {0..15}, lanes 16-31
// hold K {16..31} (contiguous, matches frag_b/frag_bT). p row-major, stride ld.
static __device__ __forceinline__ v16h frag_a(const _Float16* p, int ld) {
  int lane = threadIdx.x & 31;
  int m  = lane & 15;
  int ko = (lane >> 4) * 16;
  v16h a;
#pragma unroll
  for (int j = 0; j < 8; ++j) {
    int k = 2 * j + ko;
    a[2 * j]     = p[m * ld + k];
    a[2 * j + 1] = p[m * ld + k + 1];
  }
  return a;
}

// B fragment from K-major data: element(k,n) = p[k*ld + n].
static __device__ __forceinline__ v16h frag_b(const _Float16* p, int ld) {
  int lane = threadIdx.x & 31;
  int n  = lane & 15;
  int kb = (lane >> 4) * 16;
  v16h b;
#pragma unroll
  for (int j = 0; j < 8; ++j) {
    b[2 * j]     = p[(kb + 2 * j) * ld + n];
    b[2 * j + 1] = p[(kb + 2 * j + 1) * ld + n];
  }
  return b;
}

// B fragment used as W^T: element(k,n) = p[n*ld + k] (p row-major [N,K]).
static __device__ __forceinline__ v16h frag_bT(const _Float16* p, int ld) {
  int lane = threadIdx.x & 31;
  int n  = lane & 15;
  int kb = (lane >> 4) * 16;
  v16h b;
#pragma unroll
  for (int j = 0; j < 8; ++j) {
    b[2 * j]     = p[n * ld + kb + 2 * j];
    b[2 * j + 1] = p[n * ld + kb + 2 * j + 1];
  }
  return b;
}

static __device__ __forceinline__ float gelu_exact(float x) {
  return 0.5f * x * (1.0f + erff(x * 0.70710678118654752f));
}

// ---------------------------------------------------------------------------
// C[M,N](f32/f16) = A[M,K](f16) * W[N,K]^T(f16) (+bias)(+gelu)(+residual)
// Block: 256 thr = 8 waves -> 64x64 tile; wave grid 4(M) x 2(N); each wave
// computes 16x32 (2 WMMA sharing one A fragment). Ping-pong LDS buffers,
// async global->LDS staging double-buffered on ASYNCcnt.
// ---------------------------------------------------------------------------
template <bool GELU, bool F32OUT>
__global__ __launch_bounds__(256) void gemm_wmma_kernel(
    const _Float16* __restrict__ A, const _Float16* __restrict__ W,
    const float* __restrict__ bias, const float* __restrict__ resid,
    float* __restrict__ outf, _Float16* __restrict__ outh,
    int M, int N, int K) {
  __shared__ __align__(16) _Float16 As[2][64][32];   // 8 KB
  __shared__ __align__(16) _Float16 Ws[2][64][32];   // 8 KB
  int t    = threadIdx.x;
  int lane = t & 31;
  int w    = t >> 5;
  int m0   = blockIdx.y * 64;
  int n0   = blockIdx.x * 64;
  int mrow = (w >> 1) * 16;       // 0,16,32,48
  int ncol = (w & 1) * 32;        // 0,32
  int row  = t >> 2;              // staging: 64 rows, 4 threads/row
  int seg  = (t & 3) * 8;         // 8 halves (16B) per thread

  const _Float16* Ag = &A[(size_t)(m0 + row) * K + seg];
  const _Float16* Wg = &W[(size_t)(n0 + row) * K + seg];

  // prologue: stage tile 0 into buffer 0
  stage16(Ag, &As[0][row][seg]);
  stage16(Wg, &Ws[0][row][seg]);

  v8f acc0 = {}, acc1 = {};
  int steps = K >> 5;
  for (int i = 0; i < steps; ++i) {
    if (i + 1 < steps) {            // prefetch tile i+1 into other buffer
      int k1 = (i + 1) << 5;
      stage16(Ag + k1, &As[(i + 1) & 1][row][seg]);
      stage16(Wg + k1, &Ws[(i + 1) & 1][row][seg]);
      stage_wait<2>();              // tile i complete (in-order async loads)
    } else {
      stage_wait<0>();
    }
    __syncthreads();
    int buf = i & 1;
    v16h af  = frag_a(&As[buf][mrow][0], 32);
    v16h bf0 = frag_bT(&Ws[buf][ncol][0], 32);
    v16h bf1 = frag_bT(&Ws[buf][ncol + 16][0], 32);
    acc0 = wmma16(af, bf0, acc0);
    acc1 = wmma16(af, bf1, acc1);
    __syncthreads();
  }

  int n1 = n0 + ncol + (lane & 15);
  int n2 = n1 + 16;
  int mb = m0 + mrow + (lane >> 4) * 8;
  float bv0 = bias ? bias[n1] : 0.0f;
  float bv1 = bias ? bias[n2] : 0.0f;
#pragma unroll
  for (int r = 0; r < 8; ++r) {
    float x0 = acc0[r] + bv0;
    float x1 = acc1[r] + bv1;
    if (GELU) { x0 = gelu_exact(x0); x1 = gelu_exact(x1); }
    size_t idx = (size_t)(mb + r) * N + n1;
    if (F32OUT) {
      outf[idx]      = x0 + (resid ? resid[idx] : 0.0f);
      outf[idx + 16] = x1 + (resid ? resid[idx + 16] : 0.0f);
    } else {
      outh[idx]      = (_Float16)x0;
      outh[idx + 16] = (_Float16)x1;
    }
  }
}

// ---------------------------------------------------------------------------
// Attention: one wave per (b, h, 16-row tile). HD=32 == WMMA K exactly.
// scores (64 WMMA) -> +rpb +gbias -> softmax -> probs(f16) x V (128 WMMA).
// Dynamic LDS: 64KB f32 scores + 32KB f16 probs (gfx1250: 320KB/WGP).
// ---------------------------------------------------------------------------
__global__ __launch_bounds__(32) void attn_kernel(
    const _Float16* __restrict__ qkv,   // [B*N, 768]: q|k|v, head h at h*32
    const float* __restrict__ rpb,      // [3969, 8]
    const float* __restrict__ gb,       // [1024, 1024]
    _Float16* __restrict__ o16) {       // [B*N, 256]
  extern __shared__ char smem[];
  float*    sc = (float*)smem;                      // [16][1024]
  _Float16* pr = (_Float16*)(smem + 16 * 1024 * 4); // [16][1024]

  int lane = threadIdx.x;
  int r0 = blockIdx.x * 16;
  int h  = blockIdx.y;
  int b  = blockIdx.z;
  int m  = lane & 15;
  int mb = (lane >> 4) * 8;

  // warm L2/WGP$ for the 16 global-bias rows this tile consumes
  __builtin_prefetch((const void*)(gb + (size_t)(r0 + m) * GB_), 0, 1);

  // Q fragment: rows r0..r0+15, K = HD (reused across all 64 column chunks)
  const _Float16* qb = qkv + ((size_t)(b * N_ + r0)) * 768 + h * HD_;
  v16h qa = frag_a(qb, 768);

  // ---- scores: 16 x 1024, one WMMA per 16 key columns ----
  for (int cc = 0; cc < 64; ++cc) {
    int col0 = cc * 16;
    const _Float16* kb = qkv + ((size_t)(b * N_ + col0)) * 768 + 256 + h * HD_;
    v16h bf = frag_bT(kb, 768);   // (k^T): element(d, n) = k[col0+n][d]
    v8f s = {};
    s = wmma16(qa, bf, s);
    int kcol = col0 + m;
    int rm = kcol >> 5, cm = kcol & 31;
#pragma unroll
    for (int r = 0; r < 8; ++r) {
      int qrow = r0 + mb + r;
      int rn = qrow >> 5, cn = qrow & 31;
      int ridx = (rn - rm + 31) * 63 + (cn - cm + 31);
      sc[(mb + r) * 1024 + kcol] =
          s[r] * SCALE_ + rpb[ridx * NH_ + h] + gb[(size_t)qrow * GB_ + kcol];
    }
  }

  // ---- softmax per row (32 lanes cooperate on 1024 cols) ----
  for (int r = 0; r < 16; ++r) {
    float mx = -1e30f;
    for (int c = lane; c < 1024; c += 32) mx = fmaxf(mx, sc[r * 1024 + c]);
#pragma unroll
    for (int o = 16; o > 0; o >>= 1) mx = fmaxf(mx, __shfl_xor(mx, o, 32));
    float sum = 0.0f;
    for (int c = lane; c < 1024; c += 32) sum += __expf(sc[r * 1024 + c] - mx);
#pragma unroll
    for (int o = 16; o > 0; o >>= 1) sum += __shfl_xor(sum, o, 32);
    float inv = 1.0f / sum;
    for (int c = lane; c < 1024; c += 32)
      pr[r * 1024 + c] = (_Float16)(__expf(sc[r * 1024 + c] - mx) * inv);
  }

  // ---- out = probs[16,1024] x V[1024,32] : two 16x16 tiles per K chunk ----
  v8f acc0 = {}, acc1 = {};
  for (int c = 0; c < 64; ++c) {
    v16h pa = frag_a(pr + c * 32, 1024);
    const _Float16* vb = qkv + ((size_t)(b * N_ + c * 32)) * 768 + 512 + h * HD_;
    v16h b0 = frag_b(vb, 768);        // V natural layout: element(k,n)=v[tok][d]
    v16h b1 = frag_b(vb + 16, 768);   // d = 16..31
    acc0 = wmma16(pa, b0, acc0);
    acc1 = wmma16(pa, b1, acc1);
  }
  _Float16* ob = o16 + ((size_t)(b * N_ + r0)) * E_ + h * HD_;
#pragma unroll
  for (int r = 0; r < 8; ++r) {
    ob[(mb + r) * E_ + m]      = (_Float16)acc0[r];
    ob[(mb + r) * E_ + m + 16] = (_Float16)acc1[r];
  }
}

// ---------------------------------------------------------------------------
// LayerNorm over E=256 -> f16; one block (256 thr) per token row.
// ---------------------------------------------------------------------------
__global__ __launch_bounds__(256) void ln_kernel(
    const float* __restrict__ x, const float* __restrict__ g,
    const float* __restrict__ bb, _Float16* __restrict__ y) {
  __shared__ float red[256];
  int row = blockIdx.x;
  int e   = threadIdx.x;
  float v = x[(size_t)row * E_ + e];
  red[e] = v;
  __syncthreads();
  for (int s = 128; s > 0; s >>= 1) { if (e < s) red[e] += red[e + s]; __syncthreads(); }
  float mean = red[0] * (1.0f / 256.0f);
  __syncthreads();
  float d = v - mean;
  red[e] = d * d;
  __syncthreads();
  for (int s = 128; s > 0; s >>= 1) { if (e < s) red[e] += red[e + s]; __syncthreads(); }
  float var = red[0] * (1.0f / 256.0f);
  float rr = rsqrtf(var + 1e-5f);
  y[(size_t)row * E_ + e] = (_Float16)(d * rr * g[e] + bb[e]);
}

// ---------------------------------------------------------------------------
// Patch embed: 8x8 stride-8 conv (1 in-ch) + bias + type_embed[0]; f32 out.
// ---------------------------------------------------------------------------
__global__ __launch_bounds__(256) void patch_embed_kernel(
    const float* __restrict__ x, const float* __restrict__ pw,
    const float* __restrict__ pb, const float* __restrict__ te,
    float* __restrict__ tok) {
  __shared__ float px[64];
  int bn = blockIdx.x;          // b*1024 + n
  int b = bn >> 10, n = bn & 1023;
  int py = n >> 5, pc = n & 31;
  int t = threadIdx.x;
  if (t < 64) {
    int i = t >> 3, j = t & 7;
    px[t] = x[((size_t)b * IMG_ + py * 8 + i) * IMG_ + pc * 8 + j];
  }
  __syncthreads();
  const float* wrow = pw + t * 64;
  float acc = 0.0f;
#pragma unroll 8
  for (int q = 0; q < 64; ++q) acc += px[q] * wrow[q];
  tok[(size_t)bn * E_ + t] = acc + pb[t] + te[t];
}

// ---------------------------------------------------------------------------
// Depthwise 3x3 dilated (dil=2, pad=2) conv + GELU on [B,HID,32,32] grid.
// Activations stored token-major [B*N, HID] f16.
// ---------------------------------------------------------------------------
__global__ __launch_bounds__(256) void dwconv_kernel(
    const _Float16* __restrict__ hh, const float* __restrict__ wd,
    const float* __restrict__ bd, _Float16* __restrict__ out) {
  int idx = blockIdx.x * 256 + threadIdx.x;   // b*2^19 + n*512 + c
  int c = idx & 511;
  int n = (idx >> 9) & 1023;
  int b = idx >> 19;
  int y = n >> 5, x = n & 31;
  float acc = bd[c];
#pragma unroll
  for (int i = 0; i < 3; ++i) {
    int yy = y + (i - 1) * 2;
    if (yy < 0 || yy > 31) continue;
#pragma unroll
    for (int j = 0; j < 3; ++j) {
      int xx = x + (j - 1) * 2;
      if (xx < 0 || xx > 31) continue;
      acc += (float)hh[((size_t)(b << 10) + (yy << 5) + xx) * HID_ + c] *
             wd[c * 9 + i * 3 + j];
    }
  }
  out[(size_t)idx] = (_Float16)gelu_exact(acc);
}

__global__ __launch_bounds__(256) void f32_to_f16_kernel(
    const float* __restrict__ src, _Float16* __restrict__ dst, int n) {
  int i = blockIdx.x * 256 + threadIdx.x;
  if (i < n) dst[i] = (_Float16)src[i];
}

// ---------------------------------------------------------------------------
extern "C" void kernel_launch(void* const* d_in, const int* in_sizes, int n_in,
                              void* d_out, int out_size, void* d_ws,
                              size_t ws_size, hipStream_t stream) {
  (void)in_sizes; (void)n_in; (void)out_size; (void)ws_size;
  const float* x       = (const float*)d_in[0];
  const float* patch_w = (const float*)d_in[1];
  const float* patch_b = (const float*)d_in[2];
  const float* type_e  = (const float*)d_in[3];   // row 0 used
  const float* qkv_w   = (const float*)d_in[4];
  const float* proj_w  = (const float*)d_in[5];
  const float* proj_b  = (const float*)d_in[6];
  const float* ln1_g   = (const float*)d_in[7];
  const float* ln1_b   = (const float*)d_in[8];
  const float* ln2_g   = (const float*)d_in[9];
  const float* ln2_b   = (const float*)d_in[10];
  const float* rpb_t   = (const float*)d_in[11];  // [L,3969,8]
  const float* gbias   = (const float*)d_in[12];  // [L,1024,1024]
  const float* pw1_w   = (const float*)d_in[13];
  const float* pw1_b   = (const float*)d_in[14];
  const float* dw_w    = (const float*)d_in[15];
  const float* dw_b    = (const float*)d_in[16];
  const float* pw2_w   = (const float*)d_in[17];
  const float* pw2_b   = (const float*)d_in[18];

  // workspace layout (~54.5 MB)
  char* ws = (char*)d_ws;
  float*    tok   = (float*)(ws);                    //  8 MB f32 [8192,256]
  _Float16* y16   = (_Float16*)(ws + 8388608);       //  4 MB [8192,256]
  _Float16* qkv16 = (_Float16*)(ws + 12582912);      // 12 MB [8192,768]
  _Float16* o16   = (_Float16*)(ws + 25165824);      //  4 MB [8192,256]
  _Float16* h16   = (_Float16*)(ws + 29360128);      //  8 MB [8192,512]
  _Float16* h2    = (_Float16*)(ws + 37748736);      //  8 MB [8192,512]
  _Float16* qw16  = (_Float16*)(ws + 46137344);      // 3 MB
  _Float16* pw16  = (_Float16*)(ws + 49283072);      // 1 MB
  _Float16* w116  = (_Float16*)(ws + 50331648);      // 2 MB
  _Float16* w216  = (_Float16*)(ws + 52428800);      // 2 MB

  auto cvt = [&](const float* s, _Float16* d, int n) {
    f32_to_f16_kernel<<<(n + 255) / 256, 256, 0, stream>>>(s, d, n);
  };
  cvt(qkv_w,  qw16, L_ * 768 * 256);
  cvt(proj_w, pw16, L_ * 256 * 256);
  cvt(pw1_w,  w116, L_ * 512 * 256);
  cvt(pw2_w,  w216, L_ * 256 * 512);

  patch_embed_kernel<<<TOK_ROWS, 256, 0, stream>>>(x, patch_w, patch_b, type_e, tok);

  for (int l = 0; l < L_; ++l) {
    // attention branch
    ln_kernel<<<TOK_ROWS, 256, 0, stream>>>(tok, ln1_g + l * E_, ln1_b + l * E_, y16);
    gemm_wmma_kernel<false, false><<<dim3(768 / 64, TOK_ROWS / 64), 256, 0, stream>>>(
        y16, qw16 + (size_t)l * 768 * 256, nullptr, nullptr, nullptr, qkv16,
        TOK_ROWS, 768, 256);
    attn_kernel<<<dim3(N_ / 16, NH_, B_), 32, 98304, stream>>>(
        qkv16, rpb_t + (size_t)l * 3969 * NH_, gbias + (size_t)l * GB_ * GB_, o16);
    gemm_wmma_kernel<false, true><<<dim3(256 / 64, TOK_ROWS / 64), 256, 0, stream>>>(
        o16, pw16 + (size_t)l * 256 * 256, proj_b + l * E_, tok, tok, nullptr,
        TOK_ROWS, 256, 256);
    // conv-dilated MLP branch
    ln_kernel<<<TOK_ROWS, 256, 0, stream>>>(tok, ln2_g + l * E_, ln2_b + l * E_, y16);
    gemm_wmma_kernel<true, false><<<dim3(512 / 64, TOK_ROWS / 64), 256, 0, stream>>>(
        y16, w116 + (size_t)l * 512 * 256, pw1_b + l * HID_, nullptr, nullptr, h16,
        TOK_ROWS, 512, 256);
    dwconv_kernel<<<(TOK_ROWS * HID_) / 256, 256, 0, stream>>>(
        h16, dw_w + (size_t)l * HID_ * 9, dw_b + l * HID_, h2);
    gemm_wmma_kernel<false, true><<<dim3(256 / 64, TOK_ROWS / 64), 256, 0, stream>>>(
        h2, w216 + (size_t)l * 256 * 512, pw2_b + l * E_, tok, tok, nullptr,
        TOK_ROWS, 256, 512);
  }

  (void)hipMemcpyAsync(d_out, tok, sizeof(float) * TOK_ROWS * E_,
                       hipMemcpyDeviceToDevice, stream);
}
